// ModulatedDeformConv_41068477284904
// MI455X (gfx1250) — matrix operand order repaired and checked by reading, not compiled, not run on hardware
//
#include <hip/hip_runtime.h>
#include <hip/hip_bf16.h>

// -------- problem constants (match reference) --------
#define B_    4
#define CIN_  256
#define H_    64
#define W_    64
#define COUT_ 256
#define K_    9          // 3x3 taps
#define HO_   64
#define WO_   64
#define KC_   (CIN_ * K_)      // 2304 = GEMM K dimension, kc = c*9 + k
#define NTOT_ (B_ * HO_ * WO_) // 16384 = GEMM N dimension
#define NPX_  32               // output pixels (N columns) per block
#define NPAIR (K_ * NPX_)      // 288 (tap, pixel) pairs
#define CCH_  64               // channels staged per LDS chunk
#define KCH_  (CCH_ * K_)      // 576 kc per chunk
#define LDP_  (KCH_ + 8)       // padded LDS pitch (halfs): 584*2B=1168B, 16B-aligned rows

typedef __attribute__((ext_vector_type(16))) _Float16 v16h;
typedef __attribute__((ext_vector_type(8)))  _Float16 v8h;
typedef __attribute__((ext_vector_type(8)))  float    v8f;

union V16H { v16h v; v8h h[2]; };

__device__ __forceinline__ v8f wmma_f16(v16h a, v16h b, v8f c) {
  // D = A(16x32 f16) * B(32x16 f16) + C(16x16 f32)
  return __builtin_amdgcn_wmma_f32_16x16x32_f16(false, a, false, b,
                                                (short)0, c, false, false);
}

// -------- prologue: weight f32 -> f16 (layout [cout][c*9+k], i.e. flat copy) --------
__global__ void __launch_bounds__(256)
wconv_f16_kernel(const float* __restrict__ w, _Float16* __restrict__ wh, int n) {
  int i = blockIdx.x * 256 + threadIdx.x;
  if (i < n) wh[i] = (_Float16)w[i];
}

// -------- fused deformable-im2col (LDS) + WMMA GEMM --------
// One block per 32-pixel output tile (512 blocks). 8 waves; wave w computes
// COUT rows [32w, 32w+32) x 32 pixels as four 16x16 WMMA accumulators over
// the full K=2304, staged through LDS in 4 chunks of 576.
// Per 32-K step: 4 WMMAs <- 4 global A loads (16B each) + 4 LDS B loads.
__global__ void __launch_bounds__(256)
dcn_wmma_kernel(const float* __restrict__ x,
                const float* __restrict__ offset,
                const float* __restrict__ mask,
                const _Float16* __restrict__ wh,
                float* __restrict__ out)
{
  __shared__ __align__(16) _Float16 col[NPX_][LDP_];   // [pixel][kc_local]
  __shared__ int   s_base[NPAIR], s_o00[NPAIR], s_o01[NPAIR],
                   s_o10[NPAIR], s_o11[NPAIR];
  __shared__ float s_w00[NPAIR], s_w01[NPAIR], s_w10[NPAIR], s_w11[NPAIR];

  const int tid   = threadIdx.x;
  const int lane  = tid & 31;
  const int wave  = tid >> 5;          // 0..7
  const int nbase = blockIdx.x * NPX_; // first of 32 output pixels (flat n)

  // ---- per-(tap,pixel) bilinear coords; validity+mask folded into weights ----
  for (int pr = tid; pr < NPAIR; pr += 256) {
    const int k  = pr >> 5;            // tap 0..8
    const int p  = pr & 31;            // pixel in tile
    const int n  = nbase + p;
    const int b  = n >> 12;            // 4096 pixels per batch image
    const int hw = n & 4095;
    const int ho = hw >> 6;
    const int wo = hw & 63;
    const int ky = k / 3, kx = k - ky * 3;
    const float oy = offset[((size_t)b * (2 * K_) + 2 * k)     * 4096 + hw];
    const float ox = offset[((size_t)b * (2 * K_) + 2 * k + 1) * 4096 + hw];
    const float mv = mask  [((size_t)b * K_ + k)               * 4096 + hw];
    const float sy = (float)(ho - 1 + ky) + oy;   // STRIDE=1, PAD=1, DIL=1
    const float sx = (float)(wo - 1 + kx) + ox;
    const float fy = floorf(sy), fx = floorf(sx);
    const float wy = sy - fy,    wx = sx - fx;
    const int y0 = (int)fy, x0 = (int)fx;
    const int y1 = y0 + 1,  x1 = x0 + 1;
    const float vy0 = (y0 >= 0 && y0 < H_) ? 1.f : 0.f;
    const float vy1 = (y1 >= 0 && y1 < H_) ? 1.f : 0.f;
    const float vx0 = (x0 >= 0 && x0 < W_) ? 1.f : 0.f;
    const float vx1 = (x1 >= 0 && x1 < W_) ? 1.f : 0.f;
    const int y0c = min(max(y0, 0), H_ - 1), y1c = min(max(y1, 0), H_ - 1);
    const int x0c = min(max(x0, 0), W_ - 1), x1c = min(max(x1, 0), W_ - 1);
    s_base[pr] = b * (CIN_ * H_ * W_);
    s_o00[pr] = y0c * W_ + x0c;  s_o01[pr] = y0c * W_ + x1c;
    s_o10[pr] = y1c * W_ + x0c;  s_o11[pr] = y1c * W_ + x1c;
    s_w00[pr] = (1.f - wy) * (1.f - wx) * vy0 * vx0 * mv;
    s_w01[pr] = (1.f - wy) * wx         * vy0 * vx1 * mv;
    s_w10[pr] = wy         * (1.f - wx) * vy1 * vx0 * mv;
    s_w11[pr] = wy         * wx         * vy1 * vx1 * mv;
  }

  const int px   = tid & 31;           // pixel for sampling (coalesced x reads)
  const int cw   = tid >> 5;           // channel sub-index 0..7
  const int mrow = lane & 15;          // A row within tile / B column
  const int kgrp = lane >> 4;          // half-wave K group (0/1)

  v8f acc00 = {}, acc01 = {};          // M-tile 0 x N-tiles 0,1
  v8f acc10 = {}, acc11 = {};          // M-tile 1 x N-tiles 0,1
  const size_t arow0 = (size_t)(wave * 32 + mrow) * KC_;   // f16 weight row
  const size_t arow1 = arow0 + (size_t)16 * KC_;

  for (int cc = 0; cc < CIN_ / CCH_; ++cc) {
    __syncthreads();  // coords ready (first pass) / previous chunk consumed

    // warm next weight slice into cache while we sample
    if (cc + 1 < CIN_ / CCH_)
      __builtin_prefetch(wh + arow0 + (size_t)(cc + 1) * KCH_ + (lane << 5), 0, 1);

    // ---- fill LDS panel: 32 pixels x (64 channels * 9 taps) in f16 ----
    #pragma unroll
    for (int k = 0; k < K_; ++k) {
      const int pair = k * NPX_ + px;
      const int base = s_base[pair];
      const int o00 = s_o00[pair], o01 = s_o01[pair];
      const int o10 = s_o10[pair], o11 = s_o11[pair];
      const float w00 = s_w00[pair], w01 = s_w01[pair];
      const float w10 = s_w10[pair], w11 = s_w11[pair];
      #pragma unroll
      for (int cs = 0; cs < CCH_ / 8; ++cs) {
        const int c_local = cw + cs * 8;
        const float* xb = x + base + (size_t)(cc * CCH_ + c_local) * (H_ * W_);
        const float v = xb[o00] * w00 + xb[o01] * w01
                      + xb[o10] * w10 + xb[o11] * w11;
        col[px][c_local * K_ + k] = (_Float16)v;
      }
    }
    __syncthreads();

    // ---- WMMA over this 576-deep chunk (18 steps of K=32) ----
    const int kcg0 = cc * KCH_;
    #pragma unroll 2
    for (int kb = 0; kb < KCH_; kb += 32) {
      // B fragments from LDS: lane holds column (mrow / mrow+16),
      // contiguous K run [kb + 16*kgrp, +16)
      V16H bf0, bf1;
      const v8h* bp0 = (const v8h*)&col[mrow][kb + (kgrp << 4)];
      const v8h* bp1 = (const v8h*)&col[mrow + 16][kb + (kgrp << 4)];
      bf0.h[0] = bp0[0];  bf0.h[1] = bp0[1];
      bf1.h[0] = bp1[0];  bf1.h[1] = bp1[1];
      // A fragments from global (L2-resident f16 weights), contiguous layout:
      // lanes 0-15 hold K {kb..kb+15}; lanes 16-31 hold K {kb+16..kb+31}
      const _Float16* a0p = wh + arow0 + (size_t)(kcg0 + kb + (kgrp << 4));
      V16H af0; af0.h[0] = *(const v8h*)a0p;  af0.h[1] = *(const v8h*)(a0p + 8);
      const _Float16* a1p = wh + arow1 + (size_t)(kcg0 + kb + (kgrp << 4));
      V16H af1; af1.h[0] = *(const v8h*)a1p;  af1.h[1] = *(const v8h*)(a1p + 8);
      acc00 = wmma_f16(af0.v, bf0.v, acc00);
      acc01 = wmma_f16(af0.v, bf1.v, acc01);
      acc10 = wmma_f16(af1.v, bf0.v, acc10);
      acc11 = wmma_f16(af1.v, bf1.v, acc11);
    }
  }

  // ---- store D: VGPR j holds row M = j + 8*kgrp, column N = mrow (+16) ----
  const int n0  = nbase + mrow;        // N-tile 0 column
  const int b   = n0 >> 12;            // both tiles in same batch image
  const int hw0 = n0 & 4095;
  const int hw1 = (n0 + 16) & 4095;    // N-tile 1 column
  float* obase = out + (size_t)b * COUT_ * (HO_ * WO_);
  const int m0 = wave * 32 + 8 * kgrp;
  #pragma unroll
  for (int j = 0; j < 8; ++j) {
    obase[(size_t)(m0 + j)      * (HO_ * WO_) + hw0] = acc00[j];
    obase[(size_t)(m0 + j)      * (HO_ * WO_) + hw1] = acc01[j];
    obase[(size_t)(m0 + 16 + j) * (HO_ * WO_) + hw0] = acc10[j];
    obase[(size_t)(m0 + 16 + j) * (HO_ * WO_) + hw1] = acc11[j];
  }
}

extern "C" void kernel_launch(void* const* d_in, const int* in_sizes, int n_in,
                              void* d_out, int out_size, void* d_ws, size_t ws_size,
                              hipStream_t stream) {
  const float* x      = (const float*)d_in[0];   // [B,CIN,H,W]
  const float* offset = (const float*)d_in[1];   // [B,2K,HO,WO]
  const float* mask   = (const float*)d_in[2];   // [B,K,HO,WO]
  const float* weight = (const float*)d_in[3];   // [COUT,CIN,3,3]
  float* out = (float*)d_out;                    // [B,COUT,HO,WO]

  _Float16* wh = (_Float16*)d_ws;                // 1.18 MB f16 weights
  const int nw = COUT_ * CIN_ * K_;
  wconv_f16_kernel<<<(nw + 255) / 256, 256, 0, stream>>>(weight, wh, nw);

  const int ntiles = NTOT_ / NPX_;               // 512 blocks
  dcn_wmma_kernel<<<ntiles, 256, 0, stream>>>(x, offset, mask, wh, out);
}
